// Net_20993800143380
// MI455X (gfx1250) — matrix operand order repaired and checked by reading, not compiled, not run on hardware
//
#include <hip/hip_runtime.h>
#include <math.h>

#define HID 64
#define BN_EPS 1e-5f

typedef __attribute__((ext_vector_type(2)))  _Float16 v2h;
typedef __attribute__((ext_vector_type(8)))  _Float16 v8h;
typedef __attribute__((ext_vector_type(16))) _Float16 v16h;
typedef __attribute__((ext_vector_type(8)))  float    v8f;

#if defined(__gfx1250__) && __has_builtin(__builtin_amdgcn_wmma_f32_16x16x32_f16)
#define USE_WMMA_F16 1
#endif

__device__ __forceinline__ float eluf(float v) {
  return v > 0.0f ? v : (__expf(v) - 1.0f);
}

__global__ void fill_zero_kernel(float* __restrict__ p, int n) {
  int i = blockIdx.x * blockDim.x + threadIdx.x;
  if (i < n) p[i] = 0.0f;
}

__global__ void degree_kernel(const int* __restrict__ row, float* __restrict__ counts, int nE) {
  int e = blockIdx.x * blockDim.x + threadIdx.x;
  if (e < nE) atomicAdd(&counts[row[e]], 1.0f);
}

__global__ __launch_bounds__(128) void encoder_kernel(
    const float* __restrict__ x, const float* __restrict__ W1, const float* __restrict__ b1,
    const float* __restrict__ W2, const float* __restrict__ b2,
    float* __restrict__ h, _Float16* __restrict__ h16, int nN) {
  int node = blockIdx.x * blockDim.x + threadIdx.x;
  if (node >= nN) return;
  float xi[16];
#pragma unroll
  for (int i = 0; i < 16; ++i) xi[i] = x[(size_t)node * 16 + i];
  float t[HID];
  for (int j = 0; j < HID; ++j) {
    float acc = b1[j];
#pragma unroll
    for (int i = 0; i < 16; ++i) acc = fmaf(xi[i], W1[i * HID + j], acc);
    t[j] = eluf(acc);
  }
  for (int j = 0; j < HID; ++j) {
    float acc = b2[j];
    for (int i = 0; i < HID; ++i) acc = fmaf(t[i], W2[i * HID + j], acc);
    float hv = eluf(acc);
    h[(size_t)node * HID + j] = hv;
    h16[(size_t)node * HID + j] = (_Float16)hv;
  }
}

// Fused gather(f16) -> f16 WMMA GEMM ([16x128]@[128x64], f32 accumulate) -> bias/ELU/BN
// -> fp32 atomic scatter-add. One wave owns 16 edges; 8 waves/block share conv_W in LDS.
// Branchless: out-of-range edges write to dummy node row nN (never read back).
__global__ __launch_bounds__(256) void edge_conv_kernel(
    const _Float16* __restrict__ h16, const int* __restrict__ row, const int* __restrict__ col,
    const float* __restrict__ W,      // [128][64] (this layer's slice)
    const float* __restrict__ bias,   // [64]
    const float* __restrict__ gamma, const float* __restrict__ beta,
    const float* __restrict__ mean, const float* __restrict__ var,
    float* __restrict__ agg, int nE, int nN) {
  __shared__ __align__(16) _Float16 Wlds[64 * 128];      // column-major: Wlds[n*128 + k], 16KB
  __shared__ __align__(16) _Float16 EFlds[8][16 * 128];  // per-wave edge tile, 8 x 4KB

  const int tid = threadIdx.x;
  const int wave = tid >> 5;
  const int lane = tid & 31;

  // Stage conv_W transposed (f16) so a B fragment is two contiguous b128 LDS loads.
  for (int idx = tid; idx < 128 * 64; idx += 256) {
    int k = idx >> 6, n = idx & 63;
    Wlds[n * 128 + k] = (_Float16)W[idx];
  }

  const int ebase = blockIdx.x * 128 + wave * 16;  // 8 waves x 16 edges = 128 edges/block
  _Float16* EF = &EFlds[wave][0];
  const int c2 = lane * 2;  // 32 lanes x 2 covers 64 columns
  for (int m = 0; m < 16; ++m) {
    int e = ebase + m;
    v2h cx = {}, dn = {};
    if (e < nE) {
      int rc = row[e], cc = col[e];
      cx = *(const v2h*)&h16[(size_t)rc * HID + c2];
      v2h cn = *(const v2h*)&h16[(size_t)cc * HID + c2];
      dn = cn - cx;
    }
    *(v2h*)(&EF[m * 128 + c2])      = cx;  // ef[:,0:64]   = xc
    *(v2h*)(&EF[m * 128 + 64 + c2]) = dn;  // ef[:,64:128] = xn - xc
  }
  __syncthreads();

  const int mrow = lane & 15;
  const int hi = lane >> 4;

  // Destination rows for this lane's 8 accumulator elements (D layout: M = r + 8*hi).
  int er[8];
#pragma unroll
  for (int r = 0; r < 8; ++r) {
    int e = ebase + r + 8 * hi;
    er[r] = (e < nE) ? row[e] : nN;  // dummy row for tail edges -> no branch in epilogue
  }

  const _Float16* Arow = &EF[mrow * 128];
#if defined(USE_WMMA_F16)
  // Hoist A fragments: they depend only on the K-chunk, not the N-tile.
  // A 16x32 f16 fragment: v0-3 <- K[8hi, 8hi+8), v4-7 <- K[16+8hi, 16+8hi+8)
  v16h afrag[4];
#pragma unroll
  for (int kc = 0; kc < 4; ++kc) {
    int kb = kc * 32;
    v8h alo = *(const v8h*)(Arow + kb + 8 * hi);
    v8h ahi = *(const v8h*)(Arow + kb + 16 + 8 * hi);
    afrag[kc] = __builtin_shufflevector(alo, ahi, 0, 1, 2, 3, 4, 5, 6, 7,
                                        8, 9, 10, 11, 12, 13, 14, 15);
  }
#endif
  for (int nt = 0; nt < 4; ++nt) {
    const int ncol = nt * 16 + mrow;
    const _Float16* Bcol = &Wlds[ncol * 128];
    v8f acc = {};
#if defined(USE_WMMA_F16)
    // B 32x16 f16 fragment: v0-3 <- K[8hi, 8hi+8), v4-7 <- K[16+8hi, 16+8hi+8) at column ncol
#pragma unroll
    for (int kc = 0; kc < 4; ++kc) {
      int kb = kc * 32;
      v8h blo = *(const v8h*)(Bcol + kb + 8 * hi);
      v8h bhi = *(const v8h*)(Bcol + kb + 16 + 8 * hi);
      v16h b = __builtin_shufflevector(blo, bhi, 0, 1, 2, 3, 4, 5, 6, 7,
                                       8, 9, 10, 11, 12, 13, 14, 15);
      acc = __builtin_amdgcn_wmma_f32_16x16x32_f16(false, afrag[kc], false, b, (short)0,
                                                   acc, false, false);
    }
#else
    // Scalar fallback producing the same per-lane layout.
#pragma unroll
    for (int r = 0; r < 8; ++r) {
      const _Float16* ar = &EF[(r + 8 * hi) * 128];
      float s = 0.0f;
      for (int k = 0; k < 128; ++k) s = fmaf((float)ar[k], (float)Bcol[k], s);
      acc[r] = s;
    }
#endif
    // Folded BN after ELU: v*scale + shift
    float bb = bias[ncol];
    float sc = gamma[ncol] * rsqrtf(var[ncol] + BN_EPS);
    float sh = beta[ncol] - mean[ncol] * sc;
#pragma unroll
    for (int r = 0; r < 8; ++r) {
      float v = eluf(acc[r] + bb);
      v = fmaf(v, sc, sh);
      atomicAdd(&agg[(size_t)er[r] * HID + ncol], v);
    }
  }
}

__global__ void node_update_kernel(float* __restrict__ h, _Float16* __restrict__ h16,
                                   const float* __restrict__ agg,
                                   const float* __restrict__ counts, int total) {
  int i = blockIdx.x * blockDim.x + threadIdx.x;
  if (i < total) {
    float c = fmaxf(counts[i >> 6], 1.0f);
    float nv = h[i] + agg[i] / c;
    h[i] = nv;
    h16[i] = (_Float16)nv;
  }
}

__global__ __launch_bounds__(128) void head_kernel(
    const float* __restrict__ h, const int* __restrict__ batch,
    const float* __restrict__ Wsh, const float* __restrict__ bsh,
    const float* __restrict__ Wc1, const float* __restrict__ bc1,
    const float* __restrict__ Wc2, const float* __restrict__ bc2,
    const float* __restrict__ Ws1, const float* __restrict__ bs1,
    const float* __restrict__ sg, const float* __restrict__ sb,
    const float* __restrict__ sm, const float* __restrict__ sv,
    const float* __restrict__ Ws2, const float* __restrict__ bs2,
    const float* __restrict__ Ws3, const float* __restrict__ bs3,
    float* __restrict__ out, int nN) {
  int node = blockIdx.x * blockDim.x + threadIdx.x;
  if (node >= nN) return;
  float hv[64], feats[64];
  for (int i = 0; i < 64; ++i) hv[i] = h[(size_t)node * 64 + i];
  for (int j = 0; j < 64; ++j) {
    float acc = bsh[j];
    for (int i = 0; i < 64; ++i) acc = fmaf(hv[i], Wsh[i * 64 + j], acc);
    feats[j] = eluf(acc);
  }
  float c1[32];
  for (int j = 0; j < 32; ++j) {
    float acc = bc1[j];
    for (int i = 0; i < 64; ++i) acc = fmaf(feats[i], Wc1[i * 32 + j], acc);
    c1[j] = eluf(acc);
  }
  for (int j = 0; j < 8; ++j) {
    float acc = bc2[j];
    for (int i = 0; i < 32; ++i) acc = fmaf(c1[i], Wc2[i * 8 + j], acc);
    out[(size_t)node * 8 + j] = acc;
  }
  float s[64];
  for (int j = 0; j < 64; ++j) {
    float acc = bs1[j];
    for (int i = 0; i < 64; ++i) acc = fmaf(feats[i], Ws1[i * 64 + j], acc);
    float v = eluf(acc);
    float sc = sg[j] * rsqrtf(sv[j] + BN_EPS);
    s[j] = fmaf(v, sc, sb[j] - sm[j] * sc);
  }
  float s2[32];
  for (int j = 0; j < 32; ++j) {
    float acc = bs2[j];
    for (int i = 0; i < 64; ++i) acc = fmaf(s[i], Ws2[i * 32 + j], acc);
    s2[j] = eluf(acc);
  }
  float logit = bs3[0];
  for (int i = 0; i < 32; ++i) logit = fmaf(s2[i], Ws3[i], logit);
  out[(size_t)nN * 8 + node] = logit;
  out[(size_t)nN * 9 + node] = (float)batch[node];
}

extern "C" void kernel_launch(void* const* d_in, const int* in_sizes, int n_in,
                              void* d_out, int out_size, void* d_ws, size_t ws_size,
                              hipStream_t stream) {
  const float* x        = (const float*)d_in[0];
  const int*   ei       = (const int*)d_in[1];
  const int*   batch    = (const int*)d_in[2];
  const float* W_enc1   = (const float*)d_in[3];
  const float* b_enc1   = (const float*)d_in[4];
  const float* W_enc2   = (const float*)d_in[5];
  const float* b_enc2   = (const float*)d_in[6];
  const float* conv_W   = (const float*)d_in[7];
  const float* conv_b   = (const float*)d_in[8];
  const float* conv_g   = (const float*)d_in[9];
  const float* conv_be  = (const float*)d_in[10];
  const float* conv_m   = (const float*)d_in[11];
  const float* conv_v   = (const float*)d_in[12];
  const float* W_shared = (const float*)d_in[13];
  const float* b_shared = (const float*)d_in[14];
  const float* W_c1     = (const float*)d_in[15];
  const float* b_c1     = (const float*)d_in[16];
  const float* W_c2     = (const float*)d_in[17];
  const float* b_c2     = (const float*)d_in[18];
  const float* W_s1     = (const float*)d_in[19];
  const float* b_s1     = (const float*)d_in[20];
  const float* s_gamma  = (const float*)d_in[21];
  const float* s_beta   = (const float*)d_in[22];
  const float* s_mean   = (const float*)d_in[23];
  const float* s_var    = (const float*)d_in[24];
  const float* W_s2     = (const float*)d_in[25];
  const float* b_s2     = (const float*)d_in[26];
  const float* W_s3     = (const float*)d_in[27];
  const float* b_s3     = (const float*)d_in[28];

  const int nN = in_sizes[0] / 16;
  const int nE = in_sizes[1] / 2;
  const int* row = ei;
  const int* col = ei + nE;

  // Workspace layout: counts[nN] | h[nN*64] f32 | agg[(nN+1)*64] f32 | h16[nN*64] f16
  float* ws = (float*)d_ws;
  float* counts = ws;
  float* h      = ws + nN;
  float* agg    = h + (size_t)nN * 64;
  _Float16* h16 = (_Float16*)(agg + (size_t)(nN + 1) * 64);
  float* out    = (float*)d_out;

  fill_zero_kernel<<<(nN + 255) / 256, 256, 0, stream>>>(counts, nN);
  degree_kernel<<<(nE + 255) / 256, 256, 0, stream>>>(row, counts, nE);
  encoder_kernel<<<(nN + 127) / 128, 128, 0, stream>>>(x, W_enc1, b_enc1, W_enc2, b_enc2,
                                                       h, h16, nN);

  const int total = nN * 64;
  const int totalAgg = (nN + 1) * 64;
  for (int L = 0; L < 4; ++L) {
    fill_zero_kernel<<<(totalAgg + 255) / 256, 256, 0, stream>>>(agg, totalAgg);
    edge_conv_kernel<<<(nE + 127) / 128, 256, 0, stream>>>(
        h16, row, col,
        conv_W + (size_t)L * 128 * 64, conv_b + (size_t)L * 64,
        conv_g + (size_t)L * 64, conv_be + (size_t)L * 64,
        conv_m + (size_t)L * 64, conv_v + (size_t)L * 64,
        agg, nE, nN);
    node_update_kernel<<<(total + 255) / 256, 256, 0, stream>>>(h, h16, agg, counts, total);
  }

  head_kernel<<<(nN + 127) / 128, 128, 0, stream>>>(
      h, batch, W_shared, b_shared, W_c1, b_c1, W_c2, b_c2,
      W_s1, b_s1, s_gamma, s_beta, s_mean, s_var, W_s2, b_s2, W_s3, b_s3,
      out, nN);
}
